// GCN_89326729822741
// MI455X (gfx1250) — matrix operand order repaired and checked by reading, not compiled, not run on hardware
//
#include <hip/hip_runtime.h>
#include <hip/hip_fp16.h>

// ---------------- problem constants ----------------
constexpr int       kNumNodes = 120;
constexpr int       kB        = 1024;
constexpr int       kDeg      = 16;
constexpr int       kN        = kB * kNumNodes;           // 122880 nodes
constexpr long long kE        = (long long)kN * kDeg;     // 1966080 edges
constexpr int       kEN       = (int)(kE + kN);           // edges + self loops
constexpr int       kC1       = 40;
constexpr int       kC2       = 60;
constexpr int       kK        = kNumNodes * kC2;          // 7200
constexpr int       kOut      = 120;
constexpr int       kNPad     = 128;                      // pad 120 -> 128 for WMMA
constexpr int       kThreads  = 256;
constexpr int       kKSplit   = 5;                        // K-parallel chunks
constexpr int       kStepsPerChunk = (kK / 32) / kKSplit; // 225/5 = 45 (compile-time!)
constexpr int       kChunkK   = kStepsPerChunk * 32;      // 1440

typedef __attribute__((ext_vector_type(16))) _Float16 v16h;
typedef __attribute__((ext_vector_type(8)))  _Float16 v8h;
typedef __attribute__((ext_vector_type(8)))  float    v8f;

#define DEVFN __device__ __forceinline__

DEVFN float selu_f(float x) {
    const float sc = 1.0507009873554805f, al = 1.6732632423543772f;
    return x > 0.f ? sc * x : sc * al * (expf(x) - 1.f);
}

// ordered-int float atomic max (works for mixed signs; mbuf init = -inf)
DEVFN void atomicMaxFloat(float* addr, float val) {
    if (val >= 0.f) atomicMax((int*)addr, __float_as_int(val));
    else            atomicMin((unsigned int*)addr, (unsigned int)__float_as_int(val));
}

// ---------------- utility kernels ----------------
__global__ void fill_f32(float* p, float v, int n) {
    int i = blockIdx.x * blockDim.x + threadIdx.x;
    if (i < n) p[i] = v;
}

__global__ void reduce_sum(const float* __restrict__ v, int n, float* out) {
    __shared__ float sm[kThreads];
    float acc = 0.f;
    for (int i = blockIdx.x * blockDim.x + threadIdx.x; i < n; i += gridDim.x * blockDim.x)
        acc += v[i];
    sm[threadIdx.x] = acc;
    __syncthreads();
    for (int s = kThreads / 2; s > 0; s >>= 1) {
        if (threadIdx.x < s) sm[threadIdx.x] += sm[threadIdx.x + s];
        __syncthreads();
    }
    if (threadIdx.x == 0) atomicAdd(out, sm[0]);
}

// ---------------- GAT conv kernels ----------------
// h = x @ W  (per-node tiny GEMM), plus alpha_src/alpha_dst dot products
template <int CIN, int COUT>
__global__ void node_transform(const float* __restrict__ x, const float* __restrict__ W,
                               const float* __restrict__ att_s, const float* __restrict__ att_d,
                               float* __restrict__ h, float* __restrict__ asrc,
                               float* __restrict__ adst) {
    int i = blockIdx.x * blockDim.x + threadIdx.x;
    if (i >= kN) return;
    float xi[CIN];
#pragma unroll
    for (int k = 0; k < CIN; ++k) xi[k] = x[(size_t)i * CIN + k];
    float as = 0.f, ad = 0.f;
    for (int j = 0; j < COUT; ++j) {
        float hv = 0.f;
#pragma unroll
        for (int k = 0; k < CIN; ++k) hv += xi[k] * W[k * COUT + j];
        h[(size_t)i * COUT + j] = hv;
        as += hv * att_s[j];
        ad += hv * att_d[j];
    }
    asrc[i] = as;
    adst[i] = ad;
}

// pass 1: raw attention logit (leaky-relu), running segment max via atomics
template <int C>
__global__ void edge_pass1(const long long* __restrict__ ei, const float* __restrict__ eattr,
                           const float* __restrict__ asrc, const float* __restrict__ adst,
                           const float* __restrict__ We, const float* __restrict__ att_e,
                           const float* __restrict__ meansum, float* __restrict__ alpha,
                           float* __restrict__ mbuf) {
    int idx = blockIdx.x * blockDim.x + threadIdx.x;
    if (idx >= kEN) return;
    float c = 0.f;
    for (int k = 0; k < C; ++k) c += We[k] * att_e[k];   // (ea @ We) @ att_e, edge_dim==1
    long long s, d; float ea;
    if (idx < (int)kE) { s = ei[idx]; d = ei[kE + idx]; ea = eattr[idx]; }
    else               { s = d = idx - (int)kE; ea = meansum[0] / (float)kE; }
    float a = asrc[s] + adst[d] + ea * c;
    a = a > 0.f ? a : 0.2f * a;                           // leaky_relu 0.2
    alpha[idx] = a;
    atomicMaxFloat(mbuf + d, a);
}

// pass 2: exp(alpha - max[dst]) and segment sum
__global__ void edge_pass2(const long long* __restrict__ ei, float* __restrict__ alpha,
                           const float* __restrict__ mbuf, float* __restrict__ sbuf) {
    int idx = blockIdx.x * blockDim.x + threadIdx.x;
    if (idx >= kEN) return;
    long long d = (idx < (int)kE) ? ei[kE + idx] : (long long)(idx - (int)kE);
    float e = expf(alpha[idx] - mbuf[d]);
    alpha[idx] = e;
    atomicAdd(sbuf + d, e);
}

// pass 3: out[dst] += h[src] * e / (sum[dst] + eps)
template <int C>
__global__ void edge_pass3(const long long* __restrict__ ei, const float* __restrict__ alpha,
                           const float* __restrict__ sbuf, const float* __restrict__ h,
                           float* __restrict__ out) {
    int idx = blockIdx.x * blockDim.x + threadIdx.x;
    if (idx >= kEN) return;
    long long s, d;
    if (idx < (int)kE) { s = ei[idx]; d = ei[kE + idx]; }
    else               { s = d = idx - (int)kE; }
    float coeff = alpha[idx] / (sbuf[d] + 1e-16f);
    const float* hr = h + (size_t)s * C;
    float* orow = out + (size_t)d * C;
    for (int k = 0; k < C; ++k) atomicAdd(orow + k, hr[k] * coeff);
}

template <int C>
__global__ void bias_selu_inplace(float* __restrict__ out, const float* __restrict__ b) {
    int idx = blockIdx.x * blockDim.x + threadIdx.x;
    if (idx >= kN * C) return;
    out[idx] = selu_f(out[idx] + b[idx % C]);
}

// conv2 output: + b2, selu, convert to f16 activations for the WMMA GEMM
__global__ void bias_selu_to_f16(const float* __restrict__ in, const float* __restrict__ b,
                                 _Float16* __restrict__ out) {
    int idx = blockIdx.x * blockDim.x + threadIdx.x;
    if (idx >= kN * kC2) return;
    out[idx] = (_Float16)selu_f(in[idx] + b[idx % kC2]);
}

// Wx1 (7200x120 f32, row-major) -> transposed padded f16 (128 x 7200)
__global__ void convert_w(const float* __restrict__ Wx1, _Float16* __restrict__ wt) {
    int idx = blockIdx.x * blockDim.x + threadIdx.x;
    if (idx >= kNPad * kK) return;
    int k = idx >> 7;          // / 128
    int n = idx & (kNPad - 1); // % 128
    wt[(size_t)n * kK + k] = (n < kOut) ? (_Float16)Wx1[(size_t)k * kOut + n] : (_Float16)0.f;
}

// ---------------- WMMA GEMM: (1024 x 7200) @ (7200 x 128pad) ----------------
// Each wave32 computes a 16x64 output strip (4 N-tiles); the A fragment is
// loaded once per K-step and reused across 4 back-to-back independent WMMAs.
// K is split 5 ways (uniform 45-step compile-time chunks) across blockIdx.z,
// accumulating with global_atomic_add_f32; 64 x 2 x 5 = 640 waves in flight.
// EXPLICIT 2-deep double buffering of the A + 4 B fragments (parity-indexed
// arrays, #pragma unroll 2 so indices are compile-time and arrays live in
// registers): the 10 loads for step s+1 are issued before the 4 WMMAs of
// step s, so the s_wait lands after the WMMAs instead of stalling each one
// (round 3 emitted s_wait_loadcnt 0x0 per WMMA due to B register recycling).
// Fragment layouts (contiguous k, same in A and B): A 16x32 f16 -> lane<16 holds K
// [0..15], lane>=16 holds [16..31], row = lane&15; B 32x16 ->
// lane&15 = column, hi half takes K 16..31 (B pre-transposed so each lane's
// 16 halves are contiguous 32-byte loads).
__global__ __launch_bounds__(32) void gemm_wmma(const _Float16* __restrict__ A,
                                                const _Float16* __restrict__ Bt,
                                                float* __restrict__ out3) {
    const int lane = threadIdx.x & 31;
    const int lo = lane & 15;
    const int hi = lane >> 4;
    const int colBase = blockIdx.y * 64;
    const int kbeg = blockIdx.z * kChunkK;
    const _Float16* arow  = A  + (size_t)(blockIdx.x * 16 + lo) * kK + kbeg + hi * 16;
    const _Float16* bbase = Bt + (size_t)(colBase + lo) * kK + kbeg + hi * 16;
    const size_t bstr = (size_t)16 * kK;   // 16 columns of Bt

    v8f acc0 = {}, acc1 = {}, acc2 = {}, acc3 = {};
    union U { v16h v; v8h h[2]; };
    U a[2];
    v16h f[2][4];

    // prologue: fragments for step 0
    a[0].h[0] = *(const v8h*)(arow);
    a[0].h[1] = *(const v8h*)(arow + 8);
    f[0][0] = *(const v16h*)(bbase);
    f[0][1] = *(const v16h*)(bbase + bstr);
    f[0][2] = *(const v16h*)(bbase + 2 * bstr);
    f[0][3] = *(const v16h*)(bbase + 3 * bstr);

#pragma unroll 2
    for (int s = 0; s < kStepsPerChunk - 1; ++s) {       // 44 steady-state steps
        const int cur = s & 1, nxt = cur ^ 1;
        const int k = (s + 1) * 32;
        __builtin_prefetch(arow + k + 1024, 0, 0);       // global_prefetch_b8
        __builtin_prefetch(bbase + k + 1024, 0, 0);
        a[nxt].h[0] = *(const v8h*)(arow + k);
        a[nxt].h[1] = *(const v8h*)(arow + k + 8);
        f[nxt][0] = *(const v16h*)(bbase + k);
        f[nxt][1] = *(const v16h*)(bbase + bstr + k);
        f[nxt][2] = *(const v16h*)(bbase + 2 * bstr + k);
        f[nxt][3] = *(const v16h*)(bbase + 3 * bstr + k);
        acc0 = __builtin_amdgcn_wmma_f32_16x16x32_f16(false, a[cur].v, false, f[cur][0], (short)0, acc0, false, false);
        acc1 = __builtin_amdgcn_wmma_f32_16x16x32_f16(false, a[cur].v, false, f[cur][1], (short)0, acc1, false, false);
        acc2 = __builtin_amdgcn_wmma_f32_16x16x32_f16(false, a[cur].v, false, f[cur][2], (short)0, acc2, false, false);
        acc3 = __builtin_amdgcn_wmma_f32_16x16x32_f16(false, a[cur].v, false, f[cur][3], (short)0, acc3, false, false);
    }
    // epilogue: last step ((kStepsPerChunk-1)&1 == 0 for 45 steps)
    {
        const int cur = (kStepsPerChunk - 1) & 1;
        acc0 = __builtin_amdgcn_wmma_f32_16x16x32_f16(false, a[cur].v, false, f[cur][0], (short)0, acc0, false, false);
        acc1 = __builtin_amdgcn_wmma_f32_16x16x32_f16(false, a[cur].v, false, f[cur][1], (short)0, acc1, false, false);
        acc2 = __builtin_amdgcn_wmma_f32_16x16x32_f16(false, a[cur].v, false, f[cur][2], (short)0, acc2, false, false);
        acc3 = __builtin_amdgcn_wmma_f32_16x16x32_f16(false, a[cur].v, false, f[cur][3], (short)0, acc3, false, false);
    }
    // C/D layout: VGPR v, lanes 0-15 -> M=v, lanes 16-31 -> M=v+8; N=lane&15
    const int rbase = blockIdx.x * 16 + hi * 8;
    float* obase = out3 + (size_t)rbase * kNPad + colBase + lo;
#pragma unroll
    for (int v = 0; v < 8; ++v) {
        atomicAdd(obase + (size_t)v * kNPad + 0,  acc0[v]);
        atomicAdd(obase + (size_t)v * kNPad + 16, acc1[v]);
        atomicAdd(obase + (size_t)v * kNPad + 32, acc2[v]);
        atomicAdd(obase + (size_t)v * kNPad + 48, acc3[v]);
    }
}

// g = selu(out3[:, :120] + bx1)
__global__ void bias_selu_g(const float* __restrict__ out3, const float* __restrict__ bx1,
                            float* __restrict__ g) {
    int idx = blockIdx.x * blockDim.x + threadIdx.x;
    if (idx >= kB * kOut) return;
    int b = idx / kOut, o = idx % kOut;
    g[idx] = selu_f(out3[(size_t)b * kNPad + o] + bx1[o]);
}

// out = where(mask, g @ Wx2 + bx2, -inf)   (29 MFLOP, scalar is fine)
__global__ void final_gemm(const float* __restrict__ g, const float* __restrict__ Wx2,
                           const float* __restrict__ bx2, const unsigned char* __restrict__ mask,
                           float* __restrict__ out) {
    int idx = blockIdx.x * blockDim.x + threadIdx.x;
    if (idx >= kB * kOut) return;
    int b = idx / kOut, o = idx % kOut;
    float acc = bx2[o];
    const float* gr = g + (size_t)b * kOut;
    for (int j = 0; j < kOut; ++j) acc += gr[j] * Wx2[j * kOut + o];
    out[idx] = mask[idx] ? acc : -__builtin_inff();
}

// ---------------- host launcher ----------------
extern "C" void kernel_launch(void* const* d_in, const int* in_sizes, int n_in,
                              void* d_out, int out_size, void* d_ws, size_t ws_size,
                              hipStream_t stream) {
    (void)in_sizes; (void)n_in; (void)out_size; (void)ws_size;
    const float*         x    = (const float*)d_in[0];
    const long long*     ei   = (const long long*)d_in[1];   // int64 per reference
    const float*         eatt = (const float*)d_in[2];
    const unsigned char* mask = (const unsigned char*)d_in[3];
    const float *W1 = (const float*)d_in[4],  *as1 = (const float*)d_in[5],
                *ad1 = (const float*)d_in[6], *We1 = (const float*)d_in[7],
                *ae1 = (const float*)d_in[8], *b1  = (const float*)d_in[9];
    const float *W2 = (const float*)d_in[10], *as2 = (const float*)d_in[11],
                *ad2 = (const float*)d_in[12],*We2 = (const float*)d_in[13],
                *ae2 = (const float*)d_in[14],*b2  = (const float*)d_in[15];
    const float *Wx1 = (const float*)d_in[16], *bx1 = (const float*)d_in[17],
                *Wx2 = (const float*)d_in[18], *bx2 = (const float*)d_in[19];

    char* w = (char*)d_ws;
    size_t off = 0;
    auto alloc = [&](size_t bytes) { size_t o = off; off += (bytes + 255) & ~(size_t)255; return o; };
    const size_t H_OFF  = alloc((size_t)kN * kC2 * 4);     // h (both convs)
    const size_t A_OFF  = alloc((size_t)kN * kC2 * 4);     // conv1 out / conv2 in; later reused for f16
    const size_t B_OFF  = alloc((size_t)kN * kC2 * 4);     // conv2 out
    const size_t AS_OFF = alloc((size_t)kN * 4);
    const size_t AD_OFF = alloc((size_t)kN * 4);
    const size_t M_OFF  = alloc((size_t)kN * 4);
    const size_t S_OFF  = alloc((size_t)kN * 4);
    const size_t AL_OFF = alloc((size_t)kEN * 4);
    const size_t O3_OFF = alloc((size_t)kB * kNPad * 4);
    const size_t G_OFF  = alloc((size_t)kB * kOut * 4);
    const size_t MN_OFF = alloc(256);

    float* h    = (float*)(w + H_OFF);
    float* bufA = (float*)(w + A_OFF);
    float* bufB = (float*)(w + B_OFF);
    float* asrc = (float*)(w + AS_OFF);
    float* adst = (float*)(w + AD_OFF);
    float* mbuf = (float*)(w + M_OFF);
    float* sbuf = (float*)(w + S_OFF);
    float* alph = (float*)(w + AL_OFF);
    float* out3 = (float*)(w + O3_OFF);
    float* g    = (float*)(w + G_OFF);
    float* msum = (float*)(w + MN_OFF);
    // bufA region is dead after conv2's node_transform -> reuse for f16 buffers
    _Float16* act16 = (_Float16*)(w + A_OFF);                               // 14.75 MB
    _Float16* wt16  = (_Float16*)(w + A_OFF + (size_t)kB * kK * 2);         // +1.84 MB (fits in 29.5 MB)

    const float NEG_INF = -__builtin_huge_valf();
    auto blk = [](long long n) { return (unsigned)((n + kThreads - 1) / kThreads); };

    // -------- conv1 --------
    fill_f32<<<blk(kN), kThreads, 0, stream>>>(mbuf, NEG_INF, kN);
    fill_f32<<<blk(kN), kThreads, 0, stream>>>(sbuf, 0.f, kN);
    fill_f32<<<blk((long long)kN * kC1), kThreads, 0, stream>>>(bufA, 0.f, kN * kC1);
    fill_f32<<<1, kThreads, 0, stream>>>(msum, 0.f, 1);
    reduce_sum<<<512, kThreads, 0, stream>>>(eatt, (int)kE, msum);
    node_transform<3, kC1><<<blk(kN), kThreads, 0, stream>>>(x, W1, as1, ad1, h, asrc, adst);
    edge_pass1<kC1><<<blk(kEN), kThreads, 0, stream>>>(ei, eatt, asrc, adst, We1, ae1, msum, alph, mbuf);
    edge_pass2<<<blk(kEN), kThreads, 0, stream>>>(ei, alph, mbuf, sbuf);
    edge_pass3<kC1><<<blk(kEN), kThreads, 0, stream>>>(ei, alph, sbuf, h, bufA);
    bias_selu_inplace<kC1><<<blk((long long)kN * kC1), kThreads, 0, stream>>>(bufA, b1);

    // -------- conv2 --------
    fill_f32<<<blk(kN), kThreads, 0, stream>>>(mbuf, NEG_INF, kN);
    fill_f32<<<blk(kN), kThreads, 0, stream>>>(sbuf, 0.f, kN);
    fill_f32<<<blk((long long)kN * kC2), kThreads, 0, stream>>>(bufB, 0.f, kN * kC2);
    node_transform<kC1, kC2><<<blk(kN), kThreads, 0, stream>>>(bufA, W2, as2, ad2, h, asrc, adst);
    edge_pass1<kC2><<<blk(kEN), kThreads, 0, stream>>>(ei, eatt, asrc, adst, We2, ae2, msum, alph, mbuf);
    edge_pass2<<<blk(kEN), kThreads, 0, stream>>>(ei, alph, mbuf, sbuf);
    edge_pass3<kC2><<<blk(kEN), kThreads, 0, stream>>>(ei, alph, sbuf, h, bufB);

    // -------- MLP head --------
    bias_selu_to_f16<<<blk((long long)kN * kC2), kThreads, 0, stream>>>(bufB, b2, act16);
    convert_w<<<blk((long long)kNPad * kK), kThreads, 0, stream>>>(Wx1, wt16);
    fill_f32<<<blk((long long)kB * kNPad), kThreads, 0, stream>>>(out3, 0.f, kB * kNPad);
    gemm_wmma<<<dim3(kB / 16, kNPad / 64, kKSplit), 32, 0, stream>>>(act16, wt16, out3);
    bias_selu_g<<<blk((long long)kB * kOut), kThreads, 0, stream>>>(out3, bx1, g);
    final_gemm<<<blk((long long)kB * kOut), kThreads, 0, stream>>>(g, Wx2, bx2, mask, (float*)d_out);
}
